// Model_25563645346543
// MI455X (gfx1250) — compile-verified, matrix operand order checked by reading, not run on hardware
//
#include <hip/hip_runtime.h>

typedef __attribute__((ext_vector_type(16))) __bf16 v16bf;
typedef __attribute__((ext_vector_type(8)))  float  v8f;

#define DEV static __device__ __forceinline__

DEV unsigned short f2bf(float x) {
  union { float f; unsigned int u; } c; c.f = x;
  return (unsigned short)(c.u >> 16);
}

union BFrag { v16bf v; unsigned short u[16]; uint4 q[2]; };

DEV v8f zero8() {
  v8f z;
#pragma unroll
  for (int i = 0; i < 8; ++i) z[i] = 0.0f;
  return z;
}

DEV v8f wmma_bf16(v16bf a, v16bf b, v8f c) {
  return __builtin_amdgcn_wmma_f32_16x16x32_bf16(false, a, false, b, (short)0, c,
                                                 false, false);
}

// ---- async global->LDS staging (GLOBAL_LOAD_ASYNC_TO_LDS_B128, ASYNCcnt) ---
DEV void async_copy_f32x4(float* lds, const float* gp) {
  unsigned l = (unsigned)(unsigned long long)(void*)lds;
  asm volatile("global_load_async_to_lds_b128 %0, %1, off"
               :: "v"(l), "v"(gp) : "memory");
}
DEV void async_wait0() {
  asm volatile("s_wait_asynccnt 0" ::: "memory");
}

// ---- WMMA fragment builders -------------------------------------------------
// FAST A fragment: requires k0+32 <= K, lda % 4 == 0, 16B-aligned rows.
DEV v16bf fragA_gv(const float* __restrict__ A, int lda, int M,
                   int m0, int k0, int lane) {
  int half = (lane >> 4) & 1;
  int m = m0 + (lane & 15);
  int mc = m < M ? m : (M - 1);
  float zm = m < M ? 1.0f : 0.0f;
  const float* row = A + (long)mc * lda + k0 + 8 * half;
  float4 a0 = *(const float4*)(row + 0);
  float4 a1 = *(const float4*)(row + 4);
  float4 b0 = *(const float4*)(row + 16);
  float4 b1 = *(const float4*)(row + 20);
  BFrag f;
  f.u[0] = f2bf(a0.x * zm); f.u[1] = f2bf(a0.y * zm);
  f.u[2] = f2bf(a0.z * zm); f.u[3] = f2bf(a0.w * zm);
  f.u[4] = f2bf(a1.x * zm); f.u[5] = f2bf(a1.y * zm);
  f.u[6] = f2bf(a1.z * zm); f.u[7] = f2bf(a1.w * zm);
  f.u[8]  = f2bf(b0.x * zm); f.u[9]  = f2bf(b0.y * zm);
  f.u[10] = f2bf(b0.z * zm); f.u[11] = f2bf(b0.w * zm);
  f.u[12] = f2bf(b1.x * zm); f.u[13] = f2bf(b1.y * zm);
  f.u[14] = f2bf(b1.z * zm); f.u[15] = f2bf(b1.w * zm);
  return f.v;
}

// SLOW A fragment: branchless scalar loads (clamped index + value select).
DEV v16bf fragA_gs(const float* __restrict__ A, int lda, int M, int K,
                   int m0, int k0, int lane) {
  BFrag f;
  int half = (lane >> 4) & 1;
  int m = m0 + (lane & 15);
  int mc = m < M ? m : 0;
  float zm = m < M ? 1.0f : 0.0f;
  const float* row = A + (long)mc * lda;
#pragma unroll
  for (int i = 0; i < 8; ++i) {
    int ka = k0 + 8 * half + i;
    int kb = ka + 16;
    int kca = ka < K ? ka : 0;
    int kcb = kb < K ? kb : 0;
    float va = row[kca];
    float vb = row[kcb];
    f.u[i]     = f2bf(ka < K ? va * zm : 0.0f);
    f.u[8 + i] = f2bf(kb < K ? vb * zm : 0.0f);
  }
  return f.v;
}

// FAST B^T fragment (storage N x K row-major): same shape as fragA.
DEV v16bf fragBT_gv(const float* __restrict__ Bm, int ldb, int N,
                    int n0, int k0, int lane) {
  int half = (lane >> 4) & 1;
  int n = n0 + (lane & 15);
  int nc = n < N ? n : (N - 1);
  float zn = n < N ? 1.0f : 0.0f;
  const float* row = Bm + (long)nc * ldb + k0 + 8 * half;
  float4 a0 = *(const float4*)(row + 0);
  float4 a1 = *(const float4*)(row + 4);
  float4 b0 = *(const float4*)(row + 16);
  float4 b1 = *(const float4*)(row + 20);
  BFrag f;
  f.u[0] = f2bf(a0.x * zn); f.u[1] = f2bf(a0.y * zn);
  f.u[2] = f2bf(a0.z * zn); f.u[3] = f2bf(a0.w * zn);
  f.u[4] = f2bf(a1.x * zn); f.u[5] = f2bf(a1.y * zn);
  f.u[6] = f2bf(a1.z * zn); f.u[7] = f2bf(a1.w * zn);
  f.u[8]  = f2bf(b0.x * zn); f.u[9]  = f2bf(b0.y * zn);
  f.u[10] = f2bf(b0.z * zn); f.u[11] = f2bf(b0.w * zn);
  f.u[12] = f2bf(b1.x * zn); f.u[13] = f2bf(b1.y * zn);
  f.u[14] = f2bf(b1.z * zn); f.u[15] = f2bf(b1.w * zn);
  return f.v;
}

// SLOW B^T fragment (branchless).
DEV v16bf fragBT_gs(const float* __restrict__ Bm, int ldb, int K, int N,
                    int k0, int n0, int lane) {
  BFrag f;
  int half = (lane >> 4) & 1;
  int n = n0 + (lane & 15);
  int nc = n < N ? n : 0;
  float zn = n < N ? 1.0f : 0.0f;
  const float* row = Bm + (long)nc * ldb;
#pragma unroll
  for (int i = 0; i < 8; ++i) {
    int ka = k0 + 8 * half + i;
    int kb = ka + 16;
    int kca = ka < K ? ka : 0;
    int kcb = kb < K ? kb : 0;
    float va = row[kca];
    float vb = row[kcb];
    f.u[i]     = f2bf(ka < K ? va * zn : 0.0f);
    f.u[8 + i] = f2bf(kb < K ? vb * zn : 0.0f);
  }
  return f.v;
}

// LDS A fragment, row-major m x k (contiguous k) -> ds_load_b128 pairs.
DEV v16bf fragA_lv(const unsigned short* Sm, int lda, int m0, int k0, int lane) {
  int half = (lane >> 4) & 1;
  int m = m0 + (lane & 15);
  const unsigned short* row = Sm + m * lda + k0 + 8 * half;
  BFrag f;
  f.q[0] = *(const uint4*)(row);
  f.q[1] = *(const uint4*)(row + 16);
  return f.v;
}

// LDS B fragment from COLUMN-MAJOR tile storage: Sm[n*ldk + k] (contiguous k).
DEV v16bf fragB_lcm(const unsigned short* Sm, int ldk, int k0, int n0, int lane) {
  int half = (lane >> 4) & 1;
  int n = n0 + (lane & 15);
  const unsigned short* col = Sm + n * ldk + k0 + 8 * half;
  BFrag f;
  f.q[0] = *(const uint4*)(col);
  f.q[1] = *(const uint4*)(col + 16);
  return f.v;
}

// Store D tile (bias[m] + relu) to column-major LDS tile: one b128 per lane.
DEV void store_tile_cm_relu(unsigned short* dst, int ldk, int m0, int n0, v8f c,
                            const float* __restrict__ bias, int lane) {
  int half = (lane >> 4) & 1;
  int n = n0 + (lane & 15);
  int mb = m0 + 8 * half;
  union { uint4 q; unsigned short s[8]; } t;
#pragma unroll
  for (int r = 0; r < 8; ++r) {
    float v = c[r] + bias[mb + r];
    t.s[r] = f2bf(fmaxf(v, 0.0f));
  }
  *(uint4*)(dst + n * ldk + mb) = t.q;
}

// ---- generic batched bf16-WMMA GEMM (B always stored N x K = B^T) -----------
// Each wave owns NT consecutive 16x16 N-tiles (compile-time NT -> accumulators
// stay in named VGPRs, no movrel). One A fragment per k-step feeds NT WMMAs.
template <int NT>
__global__ void gemm_bf16_wmma(const float* __restrict__ A,
                               const float* __restrict__ Bm,
                               const float* __restrict__ bias,
                               float* __restrict__ Cm,
                               int M, int N, int K,
                               long sA, long sB, long sC,
                               int biasDim, int act, int transC) {
  int b = blockIdx.z;
  const float* Ab = A + sA * b;
  const float* Bb = Bm + sB * b;
  float* Cb = Cm + sC * b;
  int tid = threadIdx.x, wid = tid >> 5, lane = tid & 31;
  int m0 = blockIdx.y * 32 + 16 * (wid >> 2);
  int nbase = blockIdx.x * (64 * NT) + (wid & 3) * (16 * NT);
  int kFull = ((K & 3) == 0) ? (K & ~31) : 0;
  v8f acc[NT];
#pragma unroll
  for (int j = 0; j < NT; ++j) acc[j] = zero8();
  int k0 = 0;
  for (; k0 < kFull; k0 += 32) {          // vectorized fast path
    v16bf a = fragA_gv(Ab, K, M, m0, k0, lane);
#pragma unroll
    for (int j = 0; j < NT; ++j) {
      v16bf bf = fragBT_gv(Bb, K, N, nbase + 16 * j, k0, lane);
      acc[j] = wmma_bf16(a, bf, acc[j]);
    }
    if (k0 + 32 < K) {
      __builtin_prefetch(Ab + (long)(m0 + (lane & 15)) * K + k0 + 32, 0, 1);
    }
  }
  for (; k0 < K; k0 += 32) {              // branchless guarded tail
    v16bf a = fragA_gs(Ab, K, M, K, m0, k0, lane);
#pragma unroll
    for (int j = 0; j < NT; ++j) {
      v16bf bf = fragBT_gs(Bb, K, K, N, k0, nbase + 16 * j, lane);
      acc[j] = wmma_bf16(a, bf, acc[j]);
    }
  }
  int half = (lane >> 4) & 1;
#pragma unroll
  for (int j = 0; j < NT; ++j) {
    int n = nbase + 16 * j + (lane & 15);
#pragma unroll
    for (int r = 0; r < 8; ++r) {
      int m = m0 + r + 8 * half;
      if (m < M && n < N) {
        float v = acc[j][r];
        if (biasDim == 1) v += bias[m];
        else if (biasDim == 2) v += bias[n];
        if (act == 1) v = fmaxf(v, 0.0f);
        if (transC) Cb[(long)n * M + m] = v;
        else        Cb[(long)m * N + n] = v;
      }
    }
  }
}

// ---- misc small kernels -----------------------------------------------------
__global__ void prep_kernel(const float* __restrict__ x, float* __restrict__ xyz,
                            float* __restrict__ feat) {
  long g = (long)blockIdx.x * 256 + threadIdx.x;
  if (g >= 16L * 4096) return;
  const float* xi = x + g * 4;
  xyz[g * 3 + 0] = xi[0];
  xyz[g * 3 + 1] = xi[1];
  xyz[g * 3 + 2] = xi[2];
  feat[g] = xi[3];
}

__global__ void fps_kernel(const float* __restrict__ xyz, int* __restrict__ idx,
                           int N, int npoint) {
  int b = blockIdx.x, tid = threadIdx.x;
  __shared__ float dist[4096];
  __shared__ float rv[256];
  __shared__ int ri[256];
  __shared__ int s_last;
  const float* P = xyz + (long)b * N * 3;
  for (int i = tid; i < N; i += 256) dist[i] = 1e10f;
  if (tid == 0) { idx[(long)b * npoint] = 0; s_last = 0; }
  __syncthreads();
  for (int it = 1; it < npoint; ++it) {
    int last = s_last;
    float lx = P[last * 3 + 0], ly = P[last * 3 + 1], lz = P[last * 3 + 2];
    float bm = -1e30f; int bi = 0;
    for (int j = tid; j < N; j += 256) {
      float dx = P[j * 3 + 0] - lx;
      float dy = P[j * 3 + 1] - ly;
      float dz = P[j * 3 + 2] - lz;
      float d = dx * dx + dy * dy + dz * dz;
      float nd = fminf(dist[j], d);
      dist[j] = nd;
      if (nd > bm) { bm = nd; bi = j; }
    }
    rv[tid] = bm; ri[tid] = bi;
    __syncthreads();
    for (int off = 128; off > 0; off >>= 1) {
      if (tid < off) {
        if (rv[tid + off] > rv[tid] ||
            (rv[tid + off] == rv[tid] && ri[tid + off] < ri[tid])) {
          rv[tid] = rv[tid + off]; ri[tid] = ri[tid + off];
        }
      }
      __syncthreads();
    }
    if (tid == 0) { idx[(long)b * npoint + it] = ri[0]; s_last = ri[0]; }
    __syncthreads();
  }
}

__global__ void gather_xyz_kernel(const float* __restrict__ xyz,
                                  const int* __restrict__ idx,
                                  float* __restrict__ out, int N, int Snp) {
  int g = blockIdx.x * 256 + threadIdx.x;
  if (g >= 16 * Snp) return;
  int b = g / Snp;
  int id = idx[g];
  const float* p = xyz + ((long)b * N + id) * 3;
  out[g * 3 + 0] = p[0];
  out[g * 3 + 1] = p[1];
  out[g * 3 + 2] = p[2];
}

__global__ void ball_kernel(const float* __restrict__ xyz,
                            const float* __restrict__ nxyz,
                            int* __restrict__ out,
                            int N, int Snp, int ns, float r2) {
  int g = blockIdx.x * 256 + threadIdx.x;
  if (g >= 16 * Snp) return;
  int b = g / Snp;
  float cx = nxyz[g * 3 + 0], cy = nxyz[g * 3 + 1], cz = nxyz[g * 3 + 2];
  const float* P = xyz + (long)b * N * 3;
  int* O = out + (long)g * ns;
  int cnt = 0, first = -1;
  for (int n = 0; n < N && cnt < ns; ++n) {
    float dx = cx - P[n * 3 + 0];
    float dy = cy - P[n * 3 + 1];
    float dz = cz - P[n * 3 + 2];
    if (dx * dx + dy * dy + dz * dz < r2) {
      if (first < 0) first = n;
      O[cnt++] = n;
    }
  }
  if (first < 0) first = 0;
  for (; cnt < ns; ++cnt) O[cnt] = first;
}

// ---- fused SA stage 1: group(K=16) -> 4->64->64->128 MLP -> maxpool ---------
__global__ void sa1_kernel(const float* __restrict__ xyz,
                           const float* __restrict__ feat0,
                           const float* __restrict__ nxyz,
                           const int* __restrict__ ball,
                           const float* __restrict__ W1, const float* __restrict__ B1,
                           const float* __restrict__ W2, const float* __restrict__ B2,
                           const float* __restrict__ W3, const float* __restrict__ B3,
                           float* __restrict__ outFeat) {
  const int S = 1024, KP = 16, N0 = 4096;
  int s = blockIdx.x, b = blockIdx.y;
  int tid = threadIdx.x, wid = tid >> 5, lane = tid & 31;
  // column-major tiles: [point][k]
  __shared__ unsigned short g[16 * 32];
  __shared__ unsigned short o1[16 * 64];
  __shared__ unsigned short o2[16 * 64];
  __shared__ float o3[16 * 128];
  if (tid < KP) {
    for (int r = 0; r < 32; ++r) g[tid * 32 + r] = 0;
    int id = ball[((long)b * S + s) * KP + tid];
    const float* p = xyz + ((long)b * N0 + id) * 3;
    const float* c = nxyz + ((long)b * S + s) * 3;
    g[tid * 32 + 0] = f2bf(p[0] - c[0]);
    g[tid * 32 + 1] = f2bf(p[1] - c[1]);
    g[tid * 32 + 2] = f2bf(p[2] - c[2]);
    g[tid * 32 + 3] = f2bf(feat0[(long)b * N0 + id]);
  }
  __syncthreads();
  { // layer 1: (64x4)@(4x16)
    int m0 = 16 * wid;
    v16bf a = fragA_gs(W1, 4, 64, 4, m0, 0, lane);
    v16bf bb = fragB_lcm(g, 32, 0, 0, lane);
    v8f c = wmma_bf16(a, bb, zero8());
    store_tile_cm_relu(o1, 64, m0, 0, c, B1, lane);
  }
  __syncthreads();
  { // layer 2: (64x64)@(64x16)
    int m0 = 16 * wid;
    v8f c = zero8();
    for (int k0 = 0; k0 < 64; k0 += 32) {
      v16bf a = fragA_gv(W2, 64, 64, m0, k0, lane);
      v16bf bb = fragB_lcm(o1, 64, k0, 0, lane);
      c = wmma_bf16(a, bb, c);
    }
    store_tile_cm_relu(o2, 64, m0, 0, c, B2, lane);
  }
  __syncthreads();
  // layer 3: (128x64)@(64x16), column-major f32 out
  for (int t = 0; t < 2; ++t) {
    int m0 = 16 * (wid + 4 * t);
    v8f c = zero8();
    for (int k0 = 0; k0 < 64; k0 += 32) {
      v16bf a = fragA_gv(W3, 64, 128, m0, k0, lane);
      v16bf bb = fragB_lcm(o2, 64, k0, 0, lane);
      c = wmma_bf16(a, bb, c);
    }
    int half = (lane >> 4) & 1, n = lane & 15;
    int mb = m0 + 8 * half;
    float4 v0, v1;
    v0.x = fmaxf(c[0] + B3[mb + 0], 0.0f);
    v0.y = fmaxf(c[1] + B3[mb + 1], 0.0f);
    v0.z = fmaxf(c[2] + B3[mb + 2], 0.0f);
    v0.w = fmaxf(c[3] + B3[mb + 3], 0.0f);
    v1.x = fmaxf(c[4] + B3[mb + 4], 0.0f);
    v1.y = fmaxf(c[5] + B3[mb + 5], 0.0f);
    v1.z = fmaxf(c[6] + B3[mb + 6], 0.0f);
    v1.w = fmaxf(c[7] + B3[mb + 7], 0.0f);
    *(float4*)(o3 + n * 128 + mb) = v0;
    *(float4*)(o3 + n * 128 + mb + 4) = v1;
  }
  __syncthreads();
  if (tid < 128) {
    float mx = o3[tid];
    for (int k = 1; k < 16; ++k) mx = fmaxf(mx, o3[k * 128 + tid]);
    outFeat[((long)b * 128 + tid) * S + s] = mx;
  }
}

// ---- fused SA stage 2: group(K=32) -> 131->128->128->256 MLP -> maxpool -----
__global__ void sa2_kernel(const float* __restrict__ xyz1,
                           const float* __restrict__ feat1,
                           const float* __restrict__ nxyz2,
                           const int* __restrict__ ball,
                           const float* __restrict__ W1, const float* __restrict__ B1,
                           const float* __restrict__ W2, const float* __restrict__ B2,
                           const float* __restrict__ W3, const float* __restrict__ B3,
                           float* __restrict__ outFeat) {
  const int S = 256, KP = 32, N1 = 1024, CIN = 131, KPAD = 160;
  int s = blockIdx.x, b = blockIdx.y;
  int tid = threadIdx.x, wid = tid >> 5, lane = tid & 31;
  __shared__ unsigned short g[32 * KPAD];
  __shared__ unsigned short o1[32 * 128];
  __shared__ unsigned short o2[32 * 128];
  __shared__ float o3f[32 * 256];
  for (int e = tid; e < 32 * KPAD; e += 256) {
    int pnt = e / KPAD, r = e % KPAD;
    float v = 0.0f;
    if (r < CIN) {
      int id = ball[((long)b * S + s) * KP + pnt];
      if (r < 3)
        v = xyz1[((long)b * N1 + id) * 3 + r] - nxyz2[((long)b * S + s) * 3 + r];
      else
        v = feat1[((long)b * 128 + (r - 3)) * N1 + id];
    }
    g[e] = f2bf(v);
  }
  __syncthreads();
  { // layer 1: (128x131)@(131x32), K padded to 160; A shared by both n-tiles
    int m0 = 16 * wid;
    v8f c0 = zero8(), c1 = zero8();
    for (int k0 = 0; k0 < KPAD; k0 += 32) {
      v16bf a = fragA_gs(W1, CIN, 128, CIN, m0, k0, lane);
      c0 = wmma_bf16(a, fragB_lcm(g, KPAD, k0, 0, lane), c0);
      c1 = wmma_bf16(a, fragB_lcm(g, KPAD, k0, 16, lane), c1);
    }
    store_tile_cm_relu(o1, 128, m0, 0, c0, B1, lane);
    store_tile_cm_relu(o1, 128, m0, 16, c1, B1, lane);
  }
  __syncthreads();
  { // layer 2: (128x128)@(128x32)
    int m0 = 16 * wid;
    v8f c0 = zero8(), c1 = zero8();
    for (int k0 = 0; k0 < 128; k0 += 32) {
      v16bf a = fragA_gv(W2, 128, 128, m0, k0, lane);
      c0 = wmma_bf16(a, fragB_lcm(o1, 128, k0, 0, lane), c0);
      c1 = wmma_bf16(a, fragB_lcm(o1, 128, k0, 16, lane), c1);
    }
    store_tile_cm_relu(o2, 128, m0, 0, c0, B2, lane);
    store_tile_cm_relu(o2, 128, m0, 16, c1, B2, lane);
  }
  __syncthreads();
  // layer 3: (256x128)@(128x32), column-major f32 out
  for (int mt = 0; mt < 2; ++mt) {
    int m0 = 16 * (2 * wid + mt);
    v8f c0 = zero8(), c1 = zero8();
    for (int k0 = 0; k0 < 128; k0 += 32) {
      v16bf a = fragA_gv(W3, 128, 256, m0, k0, lane);
      c0 = wmma_bf16(a, fragB_lcm(o2, 128, k0, 0, lane), c0);
      c1 = wmma_bf16(a, fragB_lcm(o2, 128, k0, 16, lane), c1);
    }
    int half = (lane >> 4) & 1;
    int mb = m0 + 8 * half;
#pragma unroll
    for (int nt = 0; nt < 2; ++nt) {
      v8f cc = nt ? c1 : c0;
      int n = 16 * nt + (lane & 15);
      float4 v0, v1;
      v0.x = fmaxf(cc[0] + B3[mb + 0], 0.0f);
      v0.y = fmaxf(cc[1] + B3[mb + 1], 0.0f);
      v0.z = fmaxf(cc[2] + B3[mb + 2], 0.0f);
      v0.w = fmaxf(cc[3] + B3[mb + 3], 0.0f);
      v1.x = fmaxf(cc[4] + B3[mb + 4], 0.0f);
      v1.y = fmaxf(cc[5] + B3[mb + 5], 0.0f);
      v1.z = fmaxf(cc[6] + B3[mb + 6], 0.0f);
      v1.w = fmaxf(cc[7] + B3[mb + 7], 0.0f);
      *(float4*)(o3f + n * 256 + mb) = v0;
      *(float4*)(o3f + n * 256 + mb + 4) = v1;
    }
  }
  __syncthreads();
  {
    float mx = o3f[tid];
    for (int k = 1; k < 32; ++k) mx = fmaxf(mx, o3f[k * 256 + tid]);
    outFeat[((long)b * 256 + tid) * S + s] = mx;
  }
}

// ---- stage 3 grouping (TRANSPOSED: [point][channel]) + maxpool --------------
__global__ void g3_kernel(const float* __restrict__ nxyz2,
                          const float* __restrict__ feat2,
                          float* __restrict__ g3t) {
  long g = (long)blockIdx.x * 256 + threadIdx.x;
  long total = 16L * 256 * 259;
  if (g >= total) return;
  int b = (int)(g / (256 * 259));
  int r = (int)(g % (256 * 259));
  int n = r / 259, row = r % 259;   // g3t[(b*256+n)*259 + row]
  float v;
  if (row < 3) v = nxyz2[((long)b * 256 + n) * 3 + row];
  else v = feat2[((long)b * 256 + (row - 3)) * 256 + n];
  g3t[g] = v;
}

// input H is transposed (N=256 points x M channels); max over points.
__global__ void maxpool3_kernel(const float* __restrict__ Ht,
                                float* __restrict__ out, int M) {
  int g = blockIdx.x * 256 + threadIdx.x;
  if (g >= 16 * M) return;
  int b = g / M, m = g % M;
  const float* base = Ht + (long)b * 256 * M + m;
  float mx = base[0];
  for (int n = 1; n < 256; ++n) mx = fmaxf(mx, base[(long)n * M]);
  out[g] = mx;
}

// ---- transformer pieces -----------------------------------------------------
__global__ void transpose_cs_sc(const float* __restrict__ in,
                                float* __restrict__ out, int S, int C) {
  long g = (long)blockIdx.x * 256 + threadIdx.x;
  long total = 16L * C * S;
  if (g >= total) return;
  int b = (int)(g / ((long)C * S));
  long r = g % ((long)C * S);
  int c = (int)(r / S), s = (int)(r % S);
  out[((long)b * S + s) * C + c] = in[g];
}

__global__ void transpose_sc_cs(const float* __restrict__ in,
                                float* __restrict__ out, int S, int C) {
  long g = (long)blockIdx.x * 256 + threadIdx.x;
  long total = 16L * S * C;
  if (g >= total) return;
  int b = (int)(g / ((long)S * C));
  long r = g % ((long)S * C);
  int s = (int)(r / C), c = (int)(r % C);
  out[((long)b * C + c) * S + s] = in[g];
}

__global__ void attn_kernel(const float* __restrict__ qkv, float* __restrict__ o,
                            int S, int C, int nh) {
  int q = blockIdx.x, h = blockIdx.y, b = blockIdx.z;
  int dh = C / nh;
  int tid = threadIdx.x;
  __shared__ float sc[1024];
  __shared__ float qv[128];
  __shared__ float red[256];
  const float* base = qkv + (long)b * S * 3 * C;
  for (int d = tid; d < dh; d += 256)
    qv[d] = base[(long)q * 3 * C + h * dh + d];
  __syncthreads();
  float scale = rsqrtf((float)dh);
  float lmax = -1e30f;
  for (int j = tid; j < S; j += 256) {
    const float* kj = base + (long)j * 3 * C + C + h * dh;
    float acc = 0.0f;
    for (int d = 0; d < dh; ++d) acc += qv[d] * kj[d];
    acc *= scale;
    sc[j] = acc;
    lmax = fmaxf(lmax, acc);
  }
  red[tid] = lmax;
  __syncthreads();
  for (int off = 128; off > 0; off >>= 1) {
    if (tid < off) red[tid] = fmaxf(red[tid], red[tid + off]);
    __syncthreads();
  }
  float mx = red[0];
  __syncthreads();
  float lsum = 0.0f;
  for (int j = tid; j < S; j += 256) {
    float p = __expf(sc[j] - mx);
    sc[j] = p;
    lsum += p;
  }
  red[tid] = lsum;
  __syncthreads();
  for (int off = 128; off > 0; off >>= 1) {
    if (tid < off) red[tid] += red[tid + off];
    __syncthreads();
  }
  float inv = 1.0f / red[0];
  __syncthreads();
  for (int d = tid; d < dh; d += 256) {
    float acc = 0.0f;
    for (int j = 0; j < S; ++j)
      acc += sc[j] * base[(long)j * 3 * C + 2 * C + h * dh + d];
    o[((long)b * S + q) * C + h * dh + d] = acc * inv;
  }
}

// LN(x + r): rows staged into LDS via async global->LDS loads (ASYNCcnt).
__global__ void ln_add_kernel(const float* __restrict__ X,
                              const float* __restrict__ R,
                              const float* __restrict__ gam,
                              const float* __restrict__ bet,
                              float* __restrict__ O, int S, int C) {
  int s = blockIdx.x, b = blockIdx.y, tid = threadIdx.x;
  __shared__ float bx[512];
  __shared__ float br[512];
  __shared__ float r1[256];
  __shared__ float r2[256];
  const float* xr = X + ((long)b * S + s) * C;
  const float* rr = R + ((long)b * S + s) * C;
  float* orow = O + ((long)b * S + s) * C;
  for (int c = tid * 4; c < C; c += 1024) {
    async_copy_f32x4(&bx[c], &xr[c]);
    async_copy_f32x4(&br[c], &rr[c]);
  }
  async_wait0();
  __syncthreads();
  float sum = 0.0f, sq = 0.0f;
  for (int c = tid; c < C; c += 256) {
    float v = bx[c] + br[c];
    sum += v; sq += v * v;
  }
  r1[tid] = sum; r2[tid] = sq;
  __syncthreads();
  for (int off = 128; off > 0; off >>= 1) {
    if (tid < off) { r1[tid] += r1[tid + off]; r2[tid] += r2[tid + off]; }
    __syncthreads();
  }
  float mean = r1[0] / C;
  float var = r2[0] / C - mean * mean;
  float rstd = rsqrtf(var + 1e-5f);
  for (int c = tid; c < C; c += 256) {
    float v = bx[c] + br[c];
    orow[c] = (v - mean) * rstd * gam[c] + bet[c];
  }
}

// ---- fused FFN: out = relu(x@W1^T + b1)@W2^T + b2, hidden=2048 --------------
// k-outer loops: one A fragment per k-step feeds 8 (step1) / 4 (step2) WMMAs.
__global__ void ffn_wmma(const float* __restrict__ X,
                         const float* __restrict__ W1, const float* __restrict__ B1,
                         const float* __restrict__ W2, const float* __restrict__ B2,
                         float* __restrict__ Out, int S, int C) {
  __shared__ unsigned short h1[16 * 1024];  // 32 KB bf16 hidden chunk
  int b = blockIdx.y;
  int s0 = blockIdx.x * 16;
  int tid = threadIdx.x, wid = tid >> 5, lane = tid & 31;
  const float* Xb = X + (long)b * S * C;
  float* Ob = Out + (long)b * S * C;
  int half = (lane >> 4) & 1;
  v8f acc[4];
#pragma unroll
  for (int j = 0; j < 4; ++j) acc[j] = zero8();
  for (int hc = 0; hc < 2048; hc += 1024) {
    // step 1: h1[16 x 1024] = relu(X[s0:, :C] @ W1[hc:hc+1024, :C]^T + b1)
    v8f cc[8];
#pragma unroll
    for (int j = 0; j < 8; ++j) cc[j] = zero8();
    for (int k0 = 0; k0 < C; k0 += 32) {
      v16bf a = fragA_gv(Xb, C, S, s0, k0, lane);
#pragma unroll
      for (int j = 0; j < 8; ++j) {
        int n0h = 16 * (wid * 8 + j);
        v16bf bb = fragBT_gv(W1 + (long)hc * C, C, 1024, n0h, k0, lane);
        cc[j] = wmma_bf16(a, bb, cc[j]);
      }
    }
#pragma unroll
    for (int j = 0; j < 8; ++j) {
      int n = 16 * (wid * 8 + j) + (lane & 15);
#pragma unroll
      for (int r = 0; r < 8; ++r) {
        float v = cc[j][r] + B1[hc + n];
        h1[(r + 8 * half) * 1024 + n] = f2bf(fmaxf(v, 0.0f));
      }
    }
    __syncthreads();
    // step 2: acc += h1 @ W2[:, hc:hc+1024]^T  (A fragment shared by 4 tiles)
    for (int k0 = 0; k0 < 1024; k0 += 32) {
      v16bf a = fragA_lv(h1, 1024, 0, k0, lane);
#pragma unroll
      for (int j = 0; j < 4; ++j) {
        int n0 = 16 * (wid * 4 + j);
        v16bf bb = fragBT_gv(W2 + hc, 2048, C, n0, k0, lane);
        acc[j] = wmma_bf16(a, bb, acc[j]);
      }
    }
    __syncthreads();
  }
  for (int j = 0; j < 4; ++j) {
    int n0 = 16 * (wid * 4 + j);
    if (n0 >= C) continue;
    int n = n0 + (lane & 15);
#pragma unroll
    for (int r = 0; r < 8; ++r) {
      int srow = s0 + r + 8 * half;
      if (srow < S) Ob[(long)srow * C + n] = acc[j][r] + B2[n];
    }
  }
}

// ---- heads ------------------------------------------------------------------
__global__ void head_kernel(const float* __restrict__ W,
                            const float* __restrict__ bias,
                            const float* __restrict__ X,
                            float* __restrict__ O,
                            int Od, int Cin, int act) {
  int o = blockIdx.x * 256 + threadIdx.x;
  int b = blockIdx.y;
  if (o >= Od) return;
  const float* xr = X + (long)b * Cin;
  const float* wr = W + (long)o * Cin;
  float acc = bias[o];
  for (int c = 0; c < Cin; ++c) acc += wr[c] * xr[c];
  if (act == 1) acc = fmaxf(acc, 0.0f);
  else if (act == 2) acc = 1.0f / (1.0f + __expf(-acc));
  O[(long)b * Od + o] = acc;
}

// ============================ host side =====================================
struct TrP {
  const float *wqkv, *bqkv, *wo, *bo, *ln1s, *ln1b, *w1, *b1, *w2, *b2, *ln2s, *ln2b;
};

static void run_transformer(float* feat, int S, int C, int nh, const TrP& P,
                            float* xin, float* qkv, float* att, float* proj,
                            float* ln1, float* ffn, float* ln2,
                            hipStream_t stream) {
  const int B = 16;
  long SC = (long)S * C;
  long totalCS = (long)B * C * S;
  transpose_cs_sc<<<dim3((unsigned)((totalCS + 255) / 256)), 256, 0, stream>>>(feat, xin, S, C);
  // qkv: N = 3C >= 384 -> NT=4 (block covers 256 columns)
  gemm_bf16_wmma<4><<<dim3((3 * C + 255) / 256, (S + 31) / 32, B), 256, 0, stream>>>(
      xin, P.wqkv, P.bqkv, qkv, S, 3 * C, C, SC, 0, (long)S * 3 * C, 2, 0, 0);
  attn_kernel<<<dim3(S, nh, B), 256, 0, stream>>>(qkv, att, S, C, nh);
  if (C >= 256) {
    gemm_bf16_wmma<4><<<dim3((C + 255) / 256, (S + 31) / 32, B), 256, 0, stream>>>(
        att, P.wo, P.bo, proj, S, C, C, SC, 0, SC, 2, 0, 0);
  } else {
    gemm_bf16_wmma<2><<<dim3((C + 127) / 128, (S + 31) / 32, B), 256, 0, stream>>>(
        att, P.wo, P.bo, proj, S, C, C, SC, 0, SC, 2, 0, 0);
  }
  ln_add_kernel<<<dim3(S, B), 256, 0, stream>>>(xin, proj, P.ln1s, P.ln1b, ln1, S, C);
  ffn_wmma<<<dim3((S + 15) / 16, B), 256, 0, stream>>>(ln1, P.w1, P.b1, P.w2, P.b2, ffn, S, C);
  ln_add_kernel<<<dim3(S, B), 256, 0, stream>>>(ln1, ffn, P.ln2s, P.ln2b, ln2, S, C);
  transpose_sc_cs<<<dim3((unsigned)((totalCS + 255) / 256)), 256, 0, stream>>>(ln2, feat, S, C);
}

// workspace layout (float units)
static const long OFF_XYZ0  = 0;
static const long OFF_FEAT0 = OFF_XYZ0  + 196608;
static const long OFF_IDX1  = OFF_FEAT0 + 65536;
static const long OFF_NXYZ1 = OFF_IDX1  + 16384;
static const long OFF_BALL1 = OFF_NXYZ1 + 49152;
static const long OFF_FEAT1 = OFF_BALL1 + 262144;
static const long OFF_IDX2  = OFF_FEAT1 + 2097152;
static const long OFF_NXYZ2 = OFF_IDX2  + 4096;
static const long OFF_BALL2 = OFF_NXYZ2 + 12288;
static const long OFF_FEAT2 = OFF_BALL2 + 131072;
static const long OFF_G3    = OFF_FEAT2 + 1048576;
static const long OFF_H3A   = OFF_G3    + 1060864;
static const long OFF_H3B   = OFF_H3A   + 1048576;
static const long OFF_H3C   = OFF_H3B   + 1048576;
static const long OFF_FEAT3 = OFF_H3C   + 2097152;
static const long OFF_TXIN  = OFF_FEAT3 + 8192;
static const long OFF_TQKV  = OFF_TXIN  + 2097152;
static const long OFF_TATT  = OFF_TQKV  + 6291456;
static const long OFF_TPROJ = OFF_TATT  + 2097152;
static const long OFF_TLN1  = OFF_TPROJ + 2097152;
static const long OFF_TFFN  = OFF_TLN1  + 2097152;
static const long OFF_TLN2  = OFF_TFFN  + 2097152;
static const long OFF_HA    = OFF_TLN2  + 2097152;
static const long OFF_HB    = OFF_HA    + 4096;

extern "C" void kernel_launch(void* const* d_in, const int* in_sizes, int n_in,
                              void* d_out, int out_size, void* d_ws, size_t ws_size,
                              hipStream_t stream) {
  (void)in_sizes; (void)n_in; (void)out_size; (void)ws_size;
  const float* x = (const float*)d_in[0];
  int p = 1;
  const float *saW[3][3], *saB[3][3];
  for (int s = 0; s < 3; ++s) {
    for (int j = 0; j < 3; ++j) saW[s][j] = (const float*)d_in[p++];
    for (int j = 0; j < 3; ++j) saB[s][j] = (const float*)d_in[p++];
  }
  TrP tr[3];
  for (int s = 0; s < 3; ++s) {
    tr[s].wqkv = (const float*)d_in[p++]; tr[s].bqkv = (const float*)d_in[p++];
    tr[s].wo   = (const float*)d_in[p++]; tr[s].bo   = (const float*)d_in[p++];
    tr[s].ln1s = (const float*)d_in[p++]; tr[s].ln1b = (const float*)d_in[p++];
    tr[s].w1   = (const float*)d_in[p++]; tr[s].b1   = (const float*)d_in[p++];
    tr[s].w2   = (const float*)d_in[p++]; tr[s].b2   = (const float*)d_in[p++];
    tr[s].ln2s = (const float*)d_in[p++]; tr[s].ln2b = (const float*)d_in[p++];
  }
  const float *cW[3], *cB[3], *rW[3], *rB[3];
  for (int j = 0; j < 3; ++j) { cW[j] = (const float*)d_in[p++]; cB[j] = (const float*)d_in[p++]; }
  for (int j = 0; j < 3; ++j) { rW[j] = (const float*)d_in[p++]; rB[j] = (const float*)d_in[p++]; }

  float* ws = (float*)d_ws;
  float* xyz0  = ws + OFF_XYZ0;
  float* feat0 = ws + OFF_FEAT0;
  int*   idx1  = (int*)(ws + OFF_IDX1);
  float* nxyz1 = ws + OFF_NXYZ1;
  int*   ball1 = (int*)(ws + OFF_BALL1);
  float* feat1 = ws + OFF_FEAT1;
  int*   idx2  = (int*)(ws + OFF_IDX2);
  float* nxyz2 = ws + OFF_NXYZ2;
  int*   ball2 = (int*)(ws + OFF_BALL2);
  float* feat2 = ws + OFF_FEAT2;
  float* g3t   = ws + OFF_G3;
  float* h3a   = ws + OFF_H3A;
  float* h3b   = ws + OFF_H3B;
  float* h3c   = ws + OFF_H3C;
  float* feat3 = ws + OFF_FEAT3;
  float* txin  = ws + OFF_TXIN;
  float* tqkv  = ws + OFF_TQKV;
  float* tatt  = ws + OFF_TATT;
  float* tproj = ws + OFF_TPROJ;
  float* tln1  = ws + OFF_TLN1;
  float* tffn  = ws + OFF_TFFN;
  float* tln2  = ws + OFF_TLN2;
  float* ha    = ws + OFF_HA;
  float* hb    = ws + OFF_HB;
  float* out   = (float*)d_out;

  prep_kernel<<<dim3((16 * 4096 + 255) / 256), 256, 0, stream>>>(x, xyz0, feat0);

  // ---- stage 1 ----
  fps_kernel<<<dim3(16), 256, 0, stream>>>(xyz0, idx1, 4096, 1024);
  gather_xyz_kernel<<<dim3((16 * 1024 + 255) / 256), 256, 0, stream>>>(xyz0, idx1, nxyz1, 4096, 1024);
  ball_kernel<<<dim3((16 * 1024 + 255) / 256), 256, 0, stream>>>(xyz0, nxyz1, ball1, 4096, 1024, 16, 0.25f);
  sa1_kernel<<<dim3(1024, 16), 128, 0, stream>>>(xyz0, feat0, nxyz1, ball1,
      saW[0][0], saB[0][0], saW[0][1], saB[0][1], saW[0][2], saB[0][2], feat1);
  run_transformer(feat1, 1024, 128, 2, tr[0], txin, tqkv, tatt, tproj, tln1, tffn, tln2, stream);

  // ---- stage 2 ----
  fps_kernel<<<dim3(16), 256, 0, stream>>>(nxyz1, idx2, 1024, 256);
  gather_xyz_kernel<<<dim3((16 * 256 + 255) / 256), 256, 0, stream>>>(nxyz1, idx2, nxyz2, 1024, 256);
  ball_kernel<<<dim3((16 * 256 + 255) / 256), 256, 0, stream>>>(nxyz1, nxyz2, ball2, 1024, 256, 32, 1.0f);
  sa2_kernel<<<dim3(256, 16), 256, 0, stream>>>(nxyz1, feat1, nxyz2, ball2,
      saW[1][0], saB[1][0], saW[1][1], saB[1][1], saW[1][2], saB[1][2], feat2);
  run_transformer(feat2, 256, 256, 4, tr[1], txin, tqkv, tatt, tproj, tln1, tffn, tln2, stream);

  // ---- stage 3 (global); intermediates stored transposed [pt][ch] ----------
  g3_kernel<<<dim3((unsigned)((16L * 256 * 259 + 255) / 256)), 256, 0, stream>>>(nxyz2, feat2, g3t);
  gemm_bf16_wmma<4><<<dim3(1, 8, 16), 256, 0, stream>>>(saW[2][0], g3t, saB[2][0], h3a,
      256, 256, 259, 0, 256L * 259, 256L * 256, 1, 1, 1);
  gemm_bf16_wmma<4><<<dim3(1, 8, 16), 256, 0, stream>>>(saW[2][1], h3a, saB[2][1], h3b,
      256, 256, 256, 0, 256L * 256, 256L * 256, 1, 1, 1);
  gemm_bf16_wmma<4><<<dim3(2, 16, 16), 256, 0, stream>>>(saW[2][2], h3b, saB[2][2], h3c,
      512, 256, 256, 0, 256L * 256, 256L * 512, 1, 1, 1);
  maxpool3_kernel<<<dim3((16 * 512 + 255) / 256), 256, 0, stream>>>(h3c, feat3, 512);
  run_transformer(feat3, 1, 512, 8, tr[2], txin, tqkv, tatt, tproj, tln1, tffn, tln2, stream);

  // ---- heads ----
  head_kernel<<<dim3(1, 16), 256, 0, stream>>>(cW[0], cB[0], feat3, ha, 256, 512, 1);
  head_kernel<<<dim3(1, 16), 256, 0, stream>>>(cW[1], cB[1], ha, hb, 256, 256, 1);
  head_kernel<<<dim3(1, 16), 256, 0, stream>>>(cW[2], cB[2], hb, out + 112, 1, 256, 2);
  head_kernel<<<dim3(1, 16), 256, 0, stream>>>(rW[0], rB[0], feat3, ha, 256, 512, 1);
  head_kernel<<<dim3(1, 16), 256, 0, stream>>>(rW[1], rB[1], ha, hb, 256, 256, 1);
  head_kernel<<<dim3(1, 16), 256, 0, stream>>>(rW[2], rB[2], hb, out, 7, 256, 0);
}
